// PointConv_20255065768448
// MI455X (gfx1250) — matrix operand order repaired and checked by reading, not compiled, not run on hardware
//
#include <hip/hip_runtime.h>

typedef __attribute__((ext_vector_type(16))) _Float16 v16h;
typedef __attribute__((ext_vector_type(8)))  _Float16 v8h;
typedef __attribute__((ext_vector_type(8)))  float    v8f;
typedef __attribute__((ext_vector_type(2)))  float    v2f;

#define KNN     16
#define CMID    16
#define LASTCH  6
#define COUT    64
#define QDIM    96      /* LASTCH * CMID, c-major flatten */
#define LSTRIDE 104     /* padded k-stride (halves) for lin_sT: 208B -> conflict-free */

// wn[] LDS layout offsets (floats; all even -> float2-addressable)
#define OW1   0    /* 24  */
#define OG1   24   /* 8   */
#define OB1   32   /* 8   */
#define OW2   40   /* 64  */
#define OG2   104  /* 8   */
#define OB2   112  /* 8   */
#define OW3   120  /* 128 */
#define OG3   248  /* 16  */
#define OB3   264  /* 16  */
#define OLB   280  /* 64  */
#define WNTOT 344

__global__ __launch_bounds__(64)
void pointconv_wmma_kernel(
    const float* __restrict__ xyz,    // [N,3]
    const float* __restrict__ feats,  // [N,3]
    const int*   __restrict__ nei,    // [N,16]
    const float* __restrict__ w1, const float* __restrict__ g1, const float* __restrict__ b1,
    const float* __restrict__ w2, const float* __restrict__ g2, const float* __restrict__ b2,
    const float* __restrict__ w3, const float* __restrict__ g3, const float* __restrict__ b3,
    const float* __restrict__ lin_w,  // [96,64]
    const float* __restrict__ lin_b,  // [64]
    float* __restrict__ out_feat,     // [N,64]
    float* __restrict__ out_wni,      // [N,16,3]
    int Npts)
{
    __shared__ __align__(16) float wn[WNTOT];
    __shared__ __align__(16) _Float16 lin_sT[COUT * LSTRIDE];  // [col][k] transposed, f16
    __shared__ float4 S4[2][32][25];                           // per-wave S rows (96 used of 100)

    const int tid  = threadIdx.x;
    const int wave = tid >> 5;
    const int lane = tid & 31;

    // Warm L2/WGP$ for the shared weight matrix (global_prefetch_b8 path)
    __builtin_prefetch(lin_w + (size_t)tid * 96, 0, 0);

    // ---- stage uniform parameters into LDS (cooperative) ----
    for (int q = tid; q < 24;  q += 64) wn[OW1 + q] = w1[q];
    for (int q = tid; q < 8;   q += 64) {
        wn[OG1 + q] = g1[q]; wn[OB1 + q] = b1[q];
        wn[OG2 + q] = g2[q]; wn[OB2 + q] = b2[q];
    }
    for (int q = tid; q < 64;  q += 64) { wn[OW2 + q] = w2[q]; wn[OLB + q] = lin_b[q]; }
    for (int q = tid; q < 128; q += 64) wn[OW3 + q] = w3[q];
    for (int q = tid; q < 16;  q += 64) { wn[OG3 + q] = g3[q]; wn[OB3 + q] = b3[q]; }

    // lin_w staged TRANSPOSED as f16: lin_sT[col*LSTRIDE + k] = lin_w[k*64 + col]
    {
        const int col = tid;
#pragma unroll 1
        for (int k = 0; k < QDIM; ++k)
            lin_sT[col * LSTRIDE + k] = (_Float16)lin_w[(size_t)k * COUT + col];
    }

    const v2f* wn2 = (const v2f*)wn;   // packed-pair view of the params
    const v2f zero2 = {0.f, 0.f};

    // =========================================================
    // Phase A: one lane = one point. Gather 16 neighbors, run
    // WeightNet in packed-f32 pairs (v_pk_fma_f32 path), and
    // accumulate S[c*16+w] += featPE[c]*wgt[w] in registers.
    // =========================================================
    const int n  = blockIdx.x * 64 + tid;
    const int nc = (n < Npts) ? n : (Npts - 1);      // clamped for loads

    const float cx = xyz[3 * (size_t)nc + 0];
    const float cy = xyz[3 * (size_t)nc + 1];
    const float cz = xyz[3 * (size_t)nc + 2];

    v2f sAcc2[QDIM / 2];
#pragma unroll
    for (int q = 0; q < QDIM / 2; ++q) sAcc2[q] = zero2;

#pragma unroll 1
    for (int k = 0; k < KNN; ++k) {
        const int idx = nei[(size_t)nc * KNN + k];

        const float rx = xyz[3 * (size_t)idx + 0] - cx;
        const float ry = xyz[3 * (size_t)idx + 1] - cy;
        const float rz = xyz[3 * (size_t)idx + 2] - cz;
        const float f0 = feats[3 * (size_t)idx + 0];
        const float f1 = feats[3 * (size_t)idx + 1];
        const float f2 = feats[3 * (size_t)idx + 2];

        if (n < Npts) {
            const size_t o2 = ((size_t)n * KNN + k) * 3;
            out_wni[o2 + 0] = rx; out_wni[o2 + 1] = ry; out_wni[o2 + 2] = rz;
        }

        // WeightNet: Linear->BN(affine)->ReLU x3, two output channels per op.
        v2f h1v[4], h2v[4], wgtv[8];
#pragma unroll
        for (int p = 0; p < 4; ++p) {                 // layer 1: 3 -> 8
            v2f s = rx * wn2[OW1 / 2 + 0 + p]
                  + ry * wn2[OW1 / 2 + 4 + p]
                  + rz * wn2[OW1 / 2 + 8 + p];
            s = s * wn2[OG1 / 2 + p] + wn2[OB1 / 2 + p];
            h1v[p] = __builtin_elementwise_max(s, zero2);
        }
#pragma unroll
        for (int p = 0; p < 4; ++p) {                 // layer 2: 8 -> 8
            v2f s = zero2;
#pragma unroll
            for (int ii = 0; ii < 8; ++ii)
                s += h1v[ii >> 1][ii & 1] * wn2[OW2 / 2 + ii * 4 + p];
            s = s * wn2[OG2 / 2 + p] + wn2[OB2 / 2 + p];
            h2v[p] = __builtin_elementwise_max(s, zero2);
        }
#pragma unroll
        for (int p = 0; p < 8; ++p) {                 // layer 3: 8 -> 16
            v2f s = zero2;
#pragma unroll
            for (int ii = 0; ii < 8; ++ii)
                s += h2v[ii >> 1][ii & 1] * wn2[OW3 / 2 + ii * 8 + p];
            s = s * wn2[OG3 / 2 + p] + wn2[OB3 / 2 + p];
            wgtv[p] = __builtin_elementwise_max(s, zero2);
        }

        // Rank-1 update of S: 48 packed FMAs (96 MACs)
        const float fpe[LASTCH] = { f0, f1, f2, rx, ry, rz };
#pragma unroll
        for (int c = 0; c < LASTCH; ++c)
#pragma unroll
            for (int p = 0; p < 8; ++p)
                sAcc2[c * 8 + p] += fpe[c] * wgtv[p];
    }

    // Spill S row to LDS as 24 x b128 stores
    {
        float4* srow = &S4[wave][lane][0];
#pragma unroll
        for (int i = 0; i < 24; ++i)
            srow[i] = make_float4(sAcc2[2 * i].x,     sAcc2[2 * i].y,
                                  sAcc2[2 * i + 1].x, sAcc2[2 * i + 1].y);
    }

    __syncthreads();

    // =========================================================
    // Phase B: per wave, two M-tiles of 16 points:
    //   [16 x 96] @ [96 x 64] via 12x v_wmma_f32_16x16x32_f16 each.
    // A layout (16-bit 16x32): lane -> row M = lane&15,
    //   element j -> K = (j&7) + ((j&8)<<1) + 8*half  (contiguous octets)
    // =========================================================
    const int m    = lane & 15;
    const int half = lane >> 4;

    v16h aT[2][3];
#pragma unroll
    for (int t = 0; t < 2; ++t) {
        const float4* sr = &S4[wave][t * 16 + m][0];
#pragma unroll
        for (int ch = 0; ch < 3; ++ch) {
            const float4 x0 = sr[ch * 8 + 2 * half + 0];     // q = ch*32 + 8*half + 0..3
            const float4 x1 = sr[ch * 8 + 2 * half + 1];     // q = ... + 4..7
            const float4 x2 = sr[ch * 8 + 4 + 2 * half + 0]; // q = ch*32 + 16 + 8*half + 0..3
            const float4 x3 = sr[ch * 8 + 4 + 2 * half + 1]; // q = ... + 4..7
            v16h a;
            a[0]  = (_Float16)x0.x; a[1]  = (_Float16)x0.y; a[2]  = (_Float16)x0.z; a[3]  = (_Float16)x0.w;
            a[4]  = (_Float16)x1.x; a[5]  = (_Float16)x1.y; a[6]  = (_Float16)x1.z; a[7]  = (_Float16)x1.w;
            a[8]  = (_Float16)x2.x; a[9]  = (_Float16)x2.y; a[10] = (_Float16)x2.z; a[11] = (_Float16)x2.w;
            a[12] = (_Float16)x3.x; a[13] = (_Float16)x3.y; a[14] = (_Float16)x3.z; a[15] = (_Float16)x3.w;
            aT[t][ch] = a;
        }
    }

#pragma unroll
    for (int nt = 0; nt < 4; ++nt) {
        v8f acc0 = {}, acc1 = {};
#pragma unroll
        for (int ch = 0; ch < 3; ++ch) {
            // B layout (32x16, 16-bit): col = lane&15, element j -> K = ch*32 + (j&7) + ((j&8)<<1) + 8*half
            // -> two runs of 8 contiguous f16 (at k and k+16) in the transposed stage = 2 x ds_load_b128
            const _Float16* bp = &lin_sT[(nt * 16 + m) * LSTRIDE + ch * 32 + 8 * half];
            const v8h b0 = *(const v8h*)(bp);
            const v8h b1 = *(const v8h*)(bp + 16);
            v16h b;
#pragma unroll
            for (int j = 0; j < 8; ++j) { b[j] = b0[j]; b[8 + j] = b1[j]; }

            acc0 = __builtin_amdgcn_wmma_f32_16x16x32_f16(
                false, aT[0][ch], false, b, (short)0, acc0, false, false);
            acc1 = __builtin_amdgcn_wmma_f32_16x16x32_f16(
                false, aT[1][ch], false, b, (short)0, acc1, false, false);
        }

        // C/D layout: VGPR r -> (M = r + 8*half, N = lane&15)
        const int col = nt * 16 + m;
        const float bias = wn[OLB + col];
#pragma unroll
        for (int r = 0; r < 8; ++r) {
            const int n0 = blockIdx.x * 64 + wave * 32 + 0  + r + half * 8;
            const int n1 = blockIdx.x * 64 + wave * 32 + 16 + r + half * 8;
            if (n0 < Npts) out_feat[(size_t)n0 * COUT + col] = fmaxf(acc0[r] + bias, 0.f);
            if (n1 < Npts) out_feat[(size_t)n1 * COUT + col] = fmaxf(acc1[r] + bias, 0.f);
        }
    }
}

extern "C" void kernel_launch(void* const* d_in, const int* in_sizes, int n_in,
                              void* d_out, int out_size, void* d_ws, size_t ws_size,
                              hipStream_t stream) {
    const float* xyz   = (const float*)d_in[0];
    const float* feats = (const float*)d_in[1];
    const int*   nei   = (const int*)  d_in[2];
    const float* w1    = (const float*)d_in[3];
    const float* g1    = (const float*)d_in[4];
    const float* b1    = (const float*)d_in[5];
    const float* w2    = (const float*)d_in[6];
    const float* g2    = (const float*)d_in[7];
    const float* b2    = (const float*)d_in[8];
    const float* w3    = (const float*)d_in[9];
    const float* g3    = (const float*)d_in[10];
    const float* b3    = (const float*)d_in[11];
    const float* lin_w = (const float*)d_in[12];
    const float* lin_b = (const float*)d_in[13];

    const int Npts = in_sizes[0] / 3;

    float* out_feat = (float*)d_out;
    float* out_wni  = out_feat + (size_t)Npts * COUT;

    const int blocks = (Npts + 63) / 64;   // 64 points per block (2 waves x 32)
    pointconv_wmma_kernel<<<blocks, 64, 0, stream>>>(
        xyz, feats, nei,
        w1, g1, b1, w2, g2, b2, w3, g3, b3,
        lin_w, lin_b, out_feat, out_wni, Npts);
}
